// PraxisAttention_76501957476887
// MI455X (gfx1250) — matrix operand order repaired and checked by reading, not compiled, not run on hardware
//
#include <hip/hip_runtime.h>
#include <hip/hip_bf16.h>

// ---------------------------------------------------------------------------
// Differential attention for MI455X (gfx1250, wave32, WMMA bf16 path).
// GEMMs use double-buffered LDS tiles filled with
// GLOBAL_LOAD_ASYNC_TO_LDS_B128 (ASYNCcnt) so WMMA overlaps the panel DMA.
// ---------------------------------------------------------------------------

typedef __bf16 bf16;
typedef __attribute__((ext_vector_type(16))) __bf16 bf16x16;
typedef __attribute__((ext_vector_type(8)))  float  v8f;
typedef int v4i __attribute__((vector_size(16)));

#define Bc 2
#define Sc 2048
#define Dc 1024
#define Hc 16
#define HDc 64
#define TWO_D 2048
#define NEGBIG (-1.0e9f)

// ---- WMMA fragment loaders (layouts per CDNA5 ISA 7.12.2, wave32) ----------

// A-matrix 16x32 bf16: lanes 0-15 hold M=lane, K in {0..7, 16..23};
// lanes 16-31 hold M=lane-16, K in {8..15, 24..31}.
__device__ __forceinline__ bf16x16 load_a_frag(const bf16* row0, int stride, int lane) {
  const int m   = lane & 15;
  const int klo = (lane & 16) ? 8 : 0;
  const bf16* p = row0 + (size_t)m * stride + klo;
  bf16x16 a;
#pragma unroll
  for (int e = 0; e < 8; ++e) a[e] = p[e];
#pragma unroll
  for (int e = 0; e < 8; ++e) a[8 + e] = p[16 + e];
  return a;
}

// B-matrix 32x16 bf16: lanes 0-15 hold N=lane with K in {0..7, 16..23};
// lanes 16-31 hold N=lane-16 with K in {8..15, 24..31}.  `base` is row-major [n][k].
__device__ __forceinline__ bf16x16 load_b_frag(const bf16* base, int stride, int lane) {
  const int n  = lane & 15;
  const int k0 = (lane & 16) ? 8 : 0;
  const bf16* p = base + (size_t)n * stride + k0;
  bf16x16 b;
#pragma unroll
  for (int e = 0; e < 16; ++e) b[e] = p[(e < 8) ? e : e + 8];
  return b;
}

__device__ __forceinline__ v8f wmma_bf16(bf16x16 a, bf16x16 b, v8f c) {
  return __builtin_amdgcn_wmma_f32_16x16x32_bf16(false, a, false, b, (short)0, c,
                                                 false, false);
}

// ---- gfx1250 async LDS copy (16B per lane), guarded for toolchain drift ----
__device__ __forceinline__ void async_copy16(const bf16* g, bf16* l) {
#if __has_builtin(__builtin_amdgcn_global_load_async_to_lds_b128)
  __builtin_amdgcn_global_load_async_to_lds_b128(
      (__attribute__((address_space(1))) v4i*)g,
      (__attribute__((address_space(3))) v4i*)l, 0, 0);
#else
  *(v4i*)l = *(const v4i*)g;   // synchronous fallback
#endif
}

__device__ __forceinline__ void wait_async0() {
#if __has_builtin(__builtin_amdgcn_s_wait_asynccnt)
  __builtin_amdgcn_s_wait_asynccnt(0);
#else
  asm volatile("s_wait_asynccnt 0x0" ::: "memory");
#endif
}

// ---- small elementwise kernels ---------------------------------------------

__global__ void __launch_bounds__(256) cast_f32_bf16(const float* __restrict__ src,
                                                     bf16* __restrict__ dst, int n) {
  int i = blockIdx.x * 256 + threadIdx.x;
  if (i < n) dst[i] = (bf16)src[i];
}

__global__ void lam_kernel(const float* __restrict__ lq1, const float* __restrict__ lq2,
                           const float* __restrict__ lk1, const float* __restrict__ lk2,
                           float* __restrict__ lam) {
  if (threadIdx.x == 0) {
    float d1 = 0.f, d2 = 0.f;
    for (int i = 0; i < HDc; ++i) { d1 += lq1[i] * lk1[i]; d2 += lq2[i] * lk2[i]; }
    lam[0] = __expf(d1) - __expf(d2) + 0.8f;
  }
}

// V (B,S,D) bf16 -> Vt (B,H,HD,S) bf16 so PV B-fragments are key-contiguous.
__global__ void __launch_bounds__(256) transpose_v(const bf16* __restrict__ Vb,
                                                   bf16* __restrict__ Vt) {
  size_t i = (size_t)blockIdx.x * 256 + threadIdx.x;  // over B*S*D
  int c = (int)(i & (Dc - 1));
  size_t bs = i >> 10;
  int s = (int)(bs & (Sc - 1));
  int b = (int)(bs >> 11);
  int h = c >> 6, hd = c & 63;
  Vt[(((size_t)(b * Hc + h) * HDc + hd) * Sc) + s] = Vb[i];
}

// ---- GEMM: C[m,n] = sum_k A[m,k] * W[n,k]  (A MxK bf16, W NxK bf16) --------
// Block = 256 threads (8 waves) computing a 128x128 tile.  A/W panels
// (128x32 bf16) stream into double-buffered LDS via async-to-LDS loads;
// each wave computes 32x64 via 8 WMMAs per k-chunk.
// Requires M%128==0, N%128==0, K%32==0 (true for all uses here).
__global__ void __launch_bounds__(256)
gemm_bf16_wT(const bf16* __restrict__ A, const bf16* __restrict__ W,
             float* __restrict__ Cf, bf16* __restrict__ Cb, int M, int N, int K) {
  __shared__ bf16 As[2][128][32];
  __shared__ bf16 Ws[2][128][32];
  const int tid  = threadIdx.x;
  const int lane = tid & 31;
  const int w    = tid >> 5;
  const int wm   = w & 3;            // 4 waves over M (32 rows each)
  const int wn   = w >> 2;           // 2 waves over N (64 cols each)
  const int ntn  = N >> 7;
  const int m0   = (blockIdx.x / ntn) << 7;
  const int n0   = (blockIdx.x % ntn) << 7;

  const int srow = tid >> 1;         // staging: 2 threads per 32-elem row
  const int scol = (tid & 1) * 16;   // each thread stages 32B = 2x b128

  v8f acc[2][4] = {};

  // prologue: stage k-chunk 0 into buffer 0
  {
    const bf16* ga = A + (size_t)(m0 + srow) * K + scol;
    const bf16* gw = W + (size_t)(n0 + srow) * K + scol;
    async_copy16(ga,     &As[0][srow][scol]);
    async_copy16(ga + 8, &As[0][srow][scol + 8]);
    async_copy16(gw,     &Ws[0][srow][scol]);
    async_copy16(gw + 8, &Ws[0][srow][scol + 8]);
  }
  wait_async0();
  __syncthreads();

  const int nk = K >> 5;
  for (int kc = 0; kc < nk; ++kc) {
    const int cur = kc & 1;
    if (kc + 1 < nk) {               // stage next panel while computing
      const int k1 = (kc + 1) << 5;
      const bf16* ga = A + (size_t)(m0 + srow) * K + k1 + scol;
      const bf16* gw = W + (size_t)(n0 + srow) * K + k1 + scol;
      async_copy16(ga,     &As[cur ^ 1][srow][scol]);
      async_copy16(ga + 8, &As[cur ^ 1][srow][scol + 8]);
      async_copy16(gw,     &Ws[cur ^ 1][srow][scol]);
      async_copy16(gw + 8, &Ws[cur ^ 1][srow][scol + 8]);
    }
    bf16x16 af0 = load_a_frag(&As[cur][wm * 32][0],      32, lane);
    bf16x16 af1 = load_a_frag(&As[cur][wm * 32 + 16][0], 32, lane);
#pragma unroll
    for (int j = 0; j < 4; ++j) {
      bf16x16 bfr = load_b_frag(&Ws[cur][wn * 64 + 16 * j][0], 32, lane);
      acc[0][j] = wmma_bf16(af0, bfr, acc[0][j]);
      acc[1][j] = wmma_bf16(af1, bfr, acc[1][j]);
    }
    wait_async0();
    __syncthreads();
  }

  const int half = lane >> 4, n = lane & 15;
#pragma unroll
  for (int sub = 0; sub < 2; ++sub)
#pragma unroll
    for (int j = 0; j < 4; ++j)
#pragma unroll
      for (int r = 0; r < 8; ++r) {
        size_t idx = (size_t)(m0 + wm * 32 + sub * 16 + r + 8 * half) * N +
                     (n0 + wn * 64 + 16 * j + n);
        if (Cf) Cf[idx] = acc[sub][j][r];
        else    Cb[idx] = (bf16)acc[sub][j][r];
      }
}

// ---- Dual flash attention ---------------------------------------------------
// One wave per (b, h, 16-query tile).  Two independent online softmaxes
// (q1k1, q2k2); result = o1/l1 - lam * o2/l2, written fp32 (B,H,S,HD) plus
// deterministic per-tile (sum, sumsq) partials for the groupnorm.
__global__ void __launch_bounds__(256)
diff_flash(const bf16* __restrict__ Q, const bf16* __restrict__ Km,
           const bf16* __restrict__ Vt, const float* __restrict__ amask,
           const float* __restrict__ lam_p, float* __restrict__ Out,
           float* __restrict__ part) {
  __shared__ bf16 lds[8][16 * 32];   // per-wave P-tile staging (C->A relayout)
  const int lane = threadIdx.x & 31;
  const int wIn  = threadIdx.x >> 5;
  const int wid  = blockIdx.x * 8 + wIn;
  const int qt = wid & 127;            // S/16
  const int h  = (wid >> 7) & 15;
  const int b  = wid >> 11;
  const int q0 = qt << 4;
  const int half = lane >> 4, nn = lane & 15;
  const float lam = lam_p[0];

  const bf16* Qh = Q  + ((size_t)(b * Sc + q0)) * TWO_D + h * (2 * HDc);
  const bf16* Kh = Km + ((size_t)b * Sc) * TWO_D + h * (2 * HDc);
  const bf16* Vh = Vt + ((size_t)(b * Hc + h)) * HDc * Sc;

  v8f  o[2][4] = {};
  float mi[2][8], li[2][8];
#pragma unroll
  for (int a = 0; a < 2; ++a)
#pragma unroll
    for (int r = 0; r < 8; ++r) { mi[a][r] = -3.0e38f; li[a][r] = 0.f; }

  const int kend = q0 + 16;            // causal: keys <= q0+15
  for (int kb = 0; kb < kend; kb += 32) {
#pragma unroll
    for (int a = 0; a < 2; ++a) {
      const bf16* Qa = Qh + a * HDc;
      // --- scores: two 16x16 tiles over 32 keys, K-dim = HD = 2 WMMA chunks
      v8f s[2];
#pragma unroll
      for (int t = 0; t < 2; ++t) {
        v8f acc = {};
#pragma unroll
        for (int c = 0; c < 2; ++c) {
          bf16x16 af  = load_a_frag(Qa + 32 * c, TWO_D, lane);
          bf16x16 bfK = load_b_frag(Kh + (size_t)(kb + 16 * t) * TWO_D + a * HDc + 32 * c,
                                    TWO_D, lane);
          acc = wmma_bf16(af, bfK, acc);
        }
        s[t] = acc;
      }
      // --- scale + causal + attention mask, row max
      float rmax[8];
#pragma unroll
      for (int r = 0; r < 8; ++r) rmax[r] = -3.0e38f;
#pragma unroll
      for (int t = 0; t < 2; ++t) {
        int col = kb + 16 * t + nn;
        int colc = col < Sc ? col : (Sc - 1);
        float am = (1.0f - amask[b * Sc + colc]) * NEGBIG;
#pragma unroll
        for (int r = 0; r < 8; ++r) {
          int qi = q0 + r + 8 * half;
          float v = s[t][r] * 0.125f + am;       // 1/sqrt(64)
          if (col > qi) v = NEGBIG;
          s[t][r] = v;
          rmax[r] = fmaxf(rmax[r], v);
        }
      }
#pragma unroll
      for (int msk = 1; msk < 16; msk <<= 1)
#pragma unroll
        for (int r = 0; r < 8; ++r)
          rmax[r] = fmaxf(rmax[r], __shfl_xor(rmax[r], msk, 32));
      // --- online softmax update
      float alpha[8], rsum[8];
#pragma unroll
      for (int r = 0; r < 8; ++r) {
        float mn = fmaxf(mi[a][r], rmax[r]);
        alpha[r] = __expf(mi[a][r] - mn);
        mi[a][r] = mn;
        rsum[r] = 0.f;
      }
#pragma unroll
      for (int t = 0; t < 2; ++t)
#pragma unroll
        for (int r = 0; r < 8; ++r) {
          float p = __expf(s[t][r] - mi[a][r]);
          rsum[r] += p;
          lds[wIn][(r + 8 * half) * 32 + 16 * t + nn] = (bf16)p;
        }
#pragma unroll
      for (int msk = 1; msk < 16; msk <<= 1)
#pragma unroll
        for (int r = 0; r < 8; ++r) rsum[r] += __shfl_xor(rsum[r], msk, 32);
#pragma unroll
      for (int r = 0; r < 8; ++r) li[a][r] = li[a][r] * alpha[r] + rsum[r];
#pragma unroll
      for (int j = 0; j < 4; ++j)
#pragma unroll
        for (int r = 0; r < 8; ++r) o[a][j][r] *= alpha[r];
      // --- PV: P (16x32) from LDS as A-frag; V_t key-contiguous B-frags
      bf16x16 pa = load_a_frag(&lds[wIn][0], 32, lane);
#pragma unroll
      for (int j = 0; j < 4; ++j) {
        bf16x16 vb = load_b_frag(Vh + (size_t)(16 * j) * Sc + kb, Sc, lane);
        o[a][j] = wmma_bf16(pa, vb, o[a][j]);
      }
    }
  }
  // --- combine, store, deterministic partial sums
  float psum = 0.f, psq = 0.f;
#pragma unroll
  for (int j = 0; j < 4; ++j)
#pragma unroll
    for (int r = 0; r < 8; ++r) {
      float v = o[0][j][r] / li[0][r] - lam * (o[1][j][r] / li[1][r]);
      int m = r + 8 * half, col = 16 * j + nn;
      Out[(((size_t)(b * Hc + h) * Sc) + q0 + m) * HDc + col] = v;
      psum += v; psq += v * v;
    }
#pragma unroll
  for (int msk = 1; msk < 32; msk <<= 1) {
    psum += __shfl_xor(psum, msk, 32);
    psq  += __shfl_xor(psq,  msk, 32);
  }
  if (lane == 0) {
    int pi = ((b * Hc + h) * 128 + qt) * 2;
    part[pi] = psum; part[pi + 1] = psq;
  }
}

// ---- groupnorm stats (deterministic tree) ----------------------------------
__global__ void __launch_bounds__(128) stats_kernel(const float* __restrict__ part,
                                                    float* __restrict__ stats) {
  __shared__ float ss[128], sq[128];
  int bh = blockIdx.x, t = threadIdx.x;
  ss[t] = part[(bh * 128 + t) * 2];
  sq[t] = part[(bh * 128 + t) * 2 + 1];
  __syncthreads();
  for (int s = 64; s > 0; s >>= 1) {
    if (t < s) { ss[t] += ss[t + s]; sq[t] += sq[t + s]; }
    __syncthreads();
  }
  if (t == 0) {
    const float cnt = (float)Sc * (float)HDc;
    float mu = ss[0] / cnt;
    float var = sq[0] / cnt - mu * mu;
    stats[bh * 2] = mu;
    stats[bh * 2 + 1] = rsqrtf(var + 1e-5f);
  }
}

__global__ void __launch_bounds__(256)
norm_merge(const float* __restrict__ att, const float* __restrict__ stats,
           const float* __restrict__ gamma, const float* __restrict__ beta,
           bf16* __restrict__ merged) {
  size_t i = (size_t)blockIdx.x * 256 + threadIdx.x;   // over B*H*S*HD
  int hd = (int)(i & 63);
  size_t t = i >> 6;
  int s = (int)(t & (Sc - 1));
  int bh = (int)(t >> 11);
  int h = bh & 15, b = bh >> 4;
  float mu = stats[bh * 2], rs = stats[bh * 2 + 1];
  float x = (att[i] - mu) * rs;
  x = x * gamma[h * HDc + hd] + beta[h * HDc + hd];
  x *= 0.19999999999999996f;                            // 1 - lambda_init
  merged[((size_t)(b * Sc + s)) * Dc + h * HDc + hd] = (bf16)x;
}

// ---------------------------------------------------------------------------
extern "C" void kernel_launch(void* const* d_in, const int* in_sizes, int n_in,
                              void* d_out, int out_size, void* d_ws, size_t ws_size,
                              hipStream_t stream) {
  const float* X     = (const float*)d_in[0];
  const float* amask = (const float*)d_in[1];
  const float* Wq    = (const float*)d_in[2];
  const float* Wk    = (const float*)d_in[3];
  const float* Wv    = (const float*)d_in[4];
  const float* Wo    = (const float*)d_in[5];
  const float* lq1   = (const float*)d_in[6];
  const float* lq2   = (const float*)d_in[7];
  const float* lk1   = (const float*)d_in[8];
  const float* lk2   = (const float*)d_in[9];
  const float* gamma = (const float*)d_in[10];
  const float* beta  = (const float*)d_in[11];
  float* out = (float*)d_out;

  const size_t NX  = (size_t)Bc * Sc * Dc;     // 4,194,304
  const size_t NW2 = (size_t)2 * Dc * Dc;      // 2,097,152
  const size_t NW1 = (size_t)Dc * Dc;          // 1,048,576
  const size_t NQ  = (size_t)Bc * Sc * 2 * Dc; // 8,388,608

  // workspace layout (~96.5 MB total)
  char* w = (char*)d_ws;
  size_t off = 0;
  auto alloc = [&](size_t bytes) { char* p = w + off; off += (bytes + 255) & ~(size_t)255; return p; };
  bf16*  Xb   = (bf16*)alloc(NX * 2);
  bf16*  Wqb  = (bf16*)alloc(NW2 * 2);
  bf16*  Wkb  = (bf16*)alloc(NW2 * 2);
  bf16*  Wvb  = (bf16*)alloc(NW1 * 2);
  bf16*  Wob  = (bf16*)alloc(NW1 * 2);
  bf16*  Qb   = (bf16*)alloc(NQ * 2);
  bf16*  Kb   = (bf16*)alloc(NQ * 2);
  bf16*  Vb   = (bf16*)alloc(NX * 2);
  bf16*  Vt   = (bf16*)alloc(NX * 2);
  float* att  = (float*)alloc(NX * 4);
  bf16*  merged = (bf16*)alloc(NX * 2);
  float* part = (float*)alloc((size_t)Bc * Hc * 128 * 2 * 4);
  float* stats = (float*)alloc((size_t)Bc * Hc * 2 * 4);
  float* lam  = (float*)alloc(4);
  (void)ws_size; (void)in_sizes; (void)n_in; (void)out_size;

  // 1) fp32 -> bf16 casts
  cast_f32_bf16<<<(int)((NX  + 255) / 256), 256, 0, stream>>>(X,  Xb,  (int)NX);
  cast_f32_bf16<<<(int)((NW2 + 255) / 256), 256, 0, stream>>>(Wq, Wqb, (int)NW2);
  cast_f32_bf16<<<(int)((NW2 + 255) / 256), 256, 0, stream>>>(Wk, Wkb, (int)NW2);
  cast_f32_bf16<<<(int)((NW1 + 255) / 256), 256, 0, stream>>>(Wv, Wvb, (int)NW1);
  cast_f32_bf16<<<(int)((NW1 + 255) / 256), 256, 0, stream>>>(Wo, Wob, (int)NW1);
  lam_kernel<<<1, 32, 0, stream>>>(lq1, lq2, lk1, lk2, lam);

  // 2) projections (M=B*S=4096): 128x128 blocks of 256 threads
  const int Mrows = Bc * Sc;
  gemm_bf16_wT<<<(Mrows / 128) * (TWO_D / 128), 256, 0, stream>>>(
      Xb, Wqb, nullptr, Qb, Mrows, TWO_D, Dc);
  gemm_bf16_wT<<<(Mrows / 128) * (TWO_D / 128), 256, 0, stream>>>(
      Xb, Wkb, nullptr, Kb, Mrows, TWO_D, Dc);
  gemm_bf16_wT<<<(Mrows / 128) * (Dc / 128), 256, 0, stream>>>(
      Xb, Wvb, nullptr, Vb, Mrows, Dc, Dc);

  // 3) per-head V transpose
  transpose_v<<<(int)(NX / 256), 256, 0, stream>>>(Vb, Vt);

  // 4) dual flash attention: B*H*(S/16)=4096 waves -> 512 blocks
  diff_flash<<<(Bc * Hc * (Sc / 16)) / 8, 256, 0, stream>>>(
      Qb, Kb, Vt, amask, lam, att, part);

  // 5) groupnorm stats + normalize/merge
  stats_kernel<<<Bc * Hc, 128, 0, stream>>>(part, stats);
  norm_merge<<<(int)(NX / 256), 256, 0, stream>>>(att, stats, gamma, beta, merged);

  // 6) output projection -> fp32 d_out
  gemm_bf16_wT<<<(Mrows / 128) * (Dc / 128), 256, 0, stream>>>(
      merged, Wob, out, nullptr, Mrows, Dc, Dc);
}
